// StatementEncoder_57148834841213
// MI455X (gfx1250) — compile-verified, matrix operand order checked by reading, not run on hardware
//
#include <hip/hip_runtime.h>
#include <hip/hip_bf16.h>
#include <stdint.h>

#define NEG_SLOPE 0.2f
#define BN_EPS 1e-5f

typedef __attribute__((ext_vector_type(16))) __bf16 v16bf;
typedef __attribute__((ext_vector_type(8)))  __bf16 v8bf;
typedef __attribute__((ext_vector_type(8)))  float  v8f;

__device__ __forceinline__ unsigned short f2bf(float f){
  unsigned u = __float_as_uint(f);
  u += 0x7fffu + ((u >> 16) & 1u);          // round-to-nearest-even
  return (unsigned short)(u >> 16);
}
// monotonic float<->uint mapping for atomicMax on signed floats
__device__ __forceinline__ unsigned fenc(float f){
  unsigned u = __float_as_uint(f);
  return (u & 0x80000000u) ? ~u : (u | 0x80000000u);
}
__device__ __forceinline__ float fdec(unsigned e){
  unsigned u = (e & 0x80000000u) ? (e & 0x7fffffffu) : ~e;
  return __uint_as_float(u);
}

// ---------------- utility kernels ----------------
__global__ void k_zero(float* p, long n){
  long i = (long)blockIdx.x * blockDim.x + threadIdx.x;
  if (i < n) p[i] = 0.0f;
}

__global__ void k_cvt_bf16(const float* __restrict__ in, unsigned short* __restrict__ out, long n){
  long i = (long)blockIdx.x * blockDim.x + threadIdx.x;
  if (i < n) out[i] = f2bf(in[i]);
}

// Pack W (f32, row-major K x Ncols) into bf16 WMMA B-fragment order.
// Layout (ISA 7.12.2, 16-bit 32x16 B): lane holds col = lane&15;
// element j<8 -> k = khalf*8+j ; j>=8 -> k = 16+khalf*8+(j-8), khalf = lane>>4.
__global__ void k_pack_w(const float* __restrict__ W, unsigned short* __restrict__ out,
                         int K, int Ncols){
  int idx = blockIdx.x * blockDim.x + threadIdx.x;
  if (idx >= K * Ncols) return;
  int j      = idx & 15;
  int lane   = (idx >> 4) & 31;
  int rest   = idx >> 9;
  int Ktiles = K >> 5;
  int kt = rest % Ktiles;
  int ct = rest / Ktiles;
  int col   = ct * 16 + (lane & 15);
  int khalf = lane >> 4;
  int k = kt * 32 + ((j < 8) ? (khalf * 8 + j) : (16 + khalf * 8 + (j - 8)));
  out[idx] = f2bf(W[(size_t)k * Ncols + col]);
}

// ---------------- WMMA GEMM: C[M x Ncols] = A(bf16) @ Bpacked(bf16) ----------------
// 8 waves/block, each wave computes a 16x64 tile.
// Per block: the packed B slab (4 coltiles x Ktiles x 1KB, contiguous in memory)
// is staged into LDS once via GLOBAL_LOAD_ASYNC_TO_LDS_B128 (ASYNCcnt-tracked),
// fenced with s_wait_asynccnt + workgroup barrier, then all 8 waves feed
// v_wmma_f32_16x16x32_bf16 from LDS (ds_load_b128).
__global__ void k_gemm_bf16(const unsigned short* __restrict__ A,
                            const unsigned short* __restrict__ Bp,
                            float* __restrict__ C,
                            int K, int Ncols, int rowTiles){
  __shared__ __attribute__((aligned(16))) unsigned short smemB[24576]; // 48KB max (Ktiles<=12)

  const int wave  = threadIdx.x >> 5;
  const int lane  = threadIdx.x & 31;
  const int Ktiles = K >> 5;
  const int ct0    = blockIdx.y * 4;

  // ---- async stage of B slab into LDS (all threads participate) ----
  {
    const unsigned short* Bblk = Bp + (size_t)ct0 * Ktiles * 512; // contiguous slab
    const unsigned smemBase = (unsigned)(uintptr_t)(void*)smemB;  // LDS aperture: addr[31:0]
    const int chunks = Ktiles * 256;                              // 16B chunks, multiple of 256
    for (int i = threadIdx.x; i < chunks; i += 256){
      unsigned           lds = smemBase + (unsigned)i * 16u;
      unsigned long long g   = (unsigned long long)(uintptr_t)Bblk + (unsigned long long)i * 16ull;
      asm volatile("global_load_async_to_lds_b128 %0, %1, off"
                   :: "v"(lds), "v"(g) : "memory");
    }
    asm volatile("s_wait_asynccnt 0x0" ::: "memory");
    __syncthreads();
  }

  const int rowTile = blockIdx.x * 8 + wave;
  if (rowTile < rowTiles){                    // wave-uniform guard: EXEC all-ones for WMMA
    const int r0    = rowTile * 16;
    const int row   = r0 + (lane & 15);
    const int khalf = lane >> 4;

    v8f acc[4] = {};
    const unsigned short* Arow = A + (size_t)row * K;

    for (int kt = 0; kt < Ktiles; ++kt){
      // A fragment: row-major source, two contiguous 16B global loads per lane
      v8bf alo = *(const v8bf*)(Arow + kt * 32 + khalf * 8);
      v8bf ahi = *(const v8bf*)(Arow + kt * 32 + 16 + khalf * 8);
      v16bf a = __builtin_shufflevector(alo, ahi, 0,1,2,3,4,5,6,7,8,9,10,11,12,13,14,15);
#pragma unroll
      for (int t = 0; t < 4; ++t){
        const unsigned short* bp = smemB + ((size_t)(t * Ktiles + kt) * 32 + lane) * 16;
        v8bf blo = *(const v8bf*)bp;        // ds_load_b128
        v8bf bhi = *(const v8bf*)(bp + 8);
        v16bf b = __builtin_shufflevector(blo, bhi, 0,1,2,3,4,5,6,7,8,9,10,11,12,13,14,15);
        acc[t] = __builtin_amdgcn_wmma_f32_16x16x32_bf16(
                   false, a, false, b, (short)0, acc[t], false, false);
      }
    }
    // C/D layout: VGPR i -> row r0 + i + 8*khalf, col = tile*16 + (lane&15)
#pragma unroll
    for (int t = 0; t < 4; ++t){
      int col = (ct0 + t) * 16 + (lane & 15);
#pragma unroll
      for (int i = 0; i < 8; ++i){
        C[(size_t)(r0 + i + 8 * khalf) * Ncols + col] = acc[t][i];
      }
    }
  }
}

// ---------------- per-node attention coefficients (wave per node-head) ----------------
__global__ void k_att(const float* __restrict__ h, const float* __restrict__ aw_s,
                      const float* __restrict__ aw_d, float* __restrict__ as_o,
                      float* __restrict__ ad_o, int N, int H){
  int gid  = blockIdx.x * (blockDim.x >> 5) + (threadIdx.x >> 5);
  int lane = threadIdx.x & 31;
  if (gid >= N * H) return;
  int node = gid / H, head = gid % H;
  const float* hp = h + (size_t)node * H * 128 + head * 128;
  const float* ws = aw_s + head * 128;
  const float* wd = aw_d + head * 128;
  float s = 0.f, d = 0.f;
#pragma unroll
  for (int q = 0; q < 4; ++q){
    float v = hp[lane + 32 * q];
    s += v * ws[lane + 32 * q];
    d += v * wd[lane + 32 * q];
  }
  for (int o = 16; o > 0; o >>= 1){
    s += __shfl_xor(s, o, 32);
    d += __shfl_xor(d, o, 32);
  }
  if (lane == 0){ as_o[gid] = s; ad_o[gid] = d; }
}

// ---------------- edge passes ----------------
__global__ void k_edge_max(const long long* __restrict__ ei, long E, long ETOT,
                           const float* __restrict__ a_s, const float* __restrict__ a_d,
                           unsigned* __restrict__ m, int H){
  long e = (long)blockIdx.x * blockDim.x + threadIdx.x;
  if (e >= ETOT) return;
  long s, d;
  if (e < E){ s = ei[e]; d = ei[E + e]; } else { s = d = e - E; }   // self-loops
  for (int h = 0; h < H; ++h){
    float sc = a_s[s * H + h] + a_d[d * H + h];
    sc = sc > 0.f ? sc : NEG_SLOPE * sc;
    atomicMax(&m[d * H + h], fenc(sc));
  }
}

__global__ void k_edge_sum(const long long* __restrict__ ei, long E, long ETOT,
                           const float* __restrict__ a_s, const float* __restrict__ a_d,
                           const unsigned* __restrict__ m, float* __restrict__ z, int H){
  long e = (long)blockIdx.x * blockDim.x + threadIdx.x;
  if (e >= ETOT) return;
  long s, d;
  if (e < E){ s = ei[e]; d = ei[E + e]; } else { s = d = e - E; }
  for (int h = 0; h < H; ++h){
    float sc = a_s[s * H + h] + a_d[d * H + h];
    sc = sc > 0.f ? sc : NEG_SLOPE * sc;
    atomicAdd(&z[d * H + h], __expf(sc - fdec(m[d * H + h])));
  }
}

// wave per edge: gather h[src] (L2-resident), scatter-add alpha*h into agg[dst]
__global__ void k_edge_agg(const long long* __restrict__ ei, long E, long ETOT,
                           const float* __restrict__ a_s, const float* __restrict__ a_d,
                           const unsigned* __restrict__ m, const float* __restrict__ z,
                           const float* __restrict__ h, float* __restrict__ agg, int H){
  long gid = (long)blockIdx.x * (blockDim.x >> 5) + (threadIdx.x >> 5);
  int lane = threadIdx.x & 31;
  if (gid >= ETOT) return;
  long s, d;
  if (gid < E){ s = ei[gid]; d = ei[E + gid]; } else { s = d = gid - E; }
  float alpha = 0.f;
  if (lane < H){
    float sc = a_s[s * H + lane] + a_d[d * H + lane];
    sc = sc > 0.f ? sc : NEG_SLOPE * sc;
    alpha = __expf(sc - fdec(m[d * H + lane])) / z[d * H + lane];
  }
  const float* hs = h   + (size_t)s * H * 128;
  float*       ag = agg + (size_t)d * H * 128;
  __builtin_prefetch(hs, 0, 0);             // global_prefetch_b8 on the gathered row
  for (int hh = 0; hh < H; ++hh){
    float a = __shfl(alpha, hh, 32);
#pragma unroll
    for (int q = 0; q < 4; ++q){
      int c = hh * 128 + lane + 32 * q;
      atomicAdd(&ag[c], a * hs[c]);
    }
  }
}

// ---------------- BatchNorm (training-mode) ----------------
__global__ void k_bn_stats(const float* __restrict__ agg, const float* __restrict__ bias,
                           const float* __restrict__ gamma, const float* __restrict__ beta,
                           float* __restrict__ scale, float* __restrict__ shift,
                           int N, int C){
  int c = blockIdx.x;
  __shared__ float ss[256], s2[256];
  float a = 0.f, b = 0.f;
  for (int i = threadIdx.x; i < N; i += 256){
    float v = fmaxf(agg[(size_t)i * C + c] + bias[c], 0.f);   // relu before BN
    a += v; b += v * v;
  }
  ss[threadIdx.x] = a; s2[threadIdx.x] = b; __syncthreads();
  for (int o = 128; o > 0; o >>= 1){
    if (threadIdx.x < o){ ss[threadIdx.x] += ss[threadIdx.x + o]; s2[threadIdx.x] += s2[threadIdx.x + o]; }
    __syncthreads();
  }
  if (threadIdx.x == 0){
    float mu  = ss[0] / (float)N;
    float var = s2[0] / (float)N - mu * mu;
    float sc  = gamma[c] * rsqrtf(var + BN_EPS);
    scale[c] = sc;
    shift[c] = beta[c] - mu * sc;
  }
}

__global__ void k_bn_apply_bf16(const float* __restrict__ agg, const float* __restrict__ bias,
                                const float* __restrict__ scale, const float* __restrict__ shift,
                                unsigned short* __restrict__ out, long n, int C){
  long i = (long)blockIdx.x * blockDim.x + threadIdx.x;
  if (i >= n) return;
  int c = (int)(i % C);
  float v = fmaxf(agg[i] + bias[c], 0.f) * scale[c] + shift[c];
  out[i] = f2bf(v);
}

// ---------------- pool + MLP head ----------------
__global__ void k_colmean(const float* __restrict__ agg, const float* __restrict__ bias,
                          const float* __restrict__ scale, const float* __restrict__ shift,
                          float* __restrict__ pooled, int N, int C){
  int c = blockIdx.x;
  __shared__ float ss[256];
  float a = 0.f;
  for (int i = threadIdx.x; i < N; i += 256)
    a += fmaxf(agg[(size_t)i * C + c] + bias[c], 0.f) * scale[c] + shift[c];
  ss[threadIdx.x] = a; __syncthreads();
  for (int o = 128; o > 0; o >>= 1){
    if (threadIdx.x < o) ss[threadIdx.x] += ss[threadIdx.x + o];
    __syncthreads();
  }
  if (threadIdx.x == 0) pooled[c] = fmaxf(ss[0] / (float)N, 0.f);
}

__global__ void k_mlp(const float* __restrict__ pooled,
                      const float* __restrict__ Wm1, const float* __restrict__ bm1,
                      const float* __restrict__ Wm2, const float* __restrict__ bm2,
                      float* __restrict__ out){
  __shared__ float hid[256];
  __shared__ float pl[128];
  int t = threadIdx.x;
  if (t < 128) pl[t] = pooled[t];
  __syncthreads();
  float acc = bm1[t];
  for (int c = 0; c < 128; ++c) acc += pl[c] * Wm1[c * 256 + t];
  hid[t] = tanhf(acc);
  __syncthreads();
  if (t < 128){
    float o = bm2[t];
    for (int j = 0; j < 256; ++j) o += hid[j] * Wm2[j * 128 + t];
    out[t] = o;
  }
}

// ---------------- launcher ----------------
extern "C" void kernel_launch(void* const* d_in, const int* in_sizes, int n_in,
                              void* d_out, int out_size, void* d_ws, size_t ws_size,
                              hipStream_t stream) {
  const float*     x    = (const float*)d_in[0];
  const long long* ei   = (const long long*)d_in[1];
  const float*     W0   = (const float*)d_in[2];
  const float*     as0w = (const float*)d_in[3];
  const float*     ad0w = (const float*)d_in[4];
  const float*     b0   = (const float*)d_in[5];
  const float*     g0   = (const float*)d_in[6];
  const float*     be0  = (const float*)d_in[7];
  const float*     W1   = (const float*)d_in[8];
  const float*     as1w = (const float*)d_in[9];
  const float*     ad1w = (const float*)d_in[10];
  const float*     b1   = (const float*)d_in[11];
  const float*     g1   = (const float*)d_in[12];
  const float*     be1  = (const float*)d_in[13];
  const float*     Wm1  = (const float*)d_in[14];
  const float*     bm1  = (const float*)d_in[15];
  const float*     Wm2  = (const float*)d_in[16];
  const float*     bm2  = (const float*)d_in[17];

  const int  N    = in_sizes[0] / 128;         // 50000
  const long E    = in_sizes[1] / 2;           // 800000
  const long ETOT = E + (long)N;               // + self-loops

  // carve workspace (256B aligned)
  unsigned char* ws = (unsigned char*)d_ws;
  size_t off = 0;
  auto carve = [&](size_t bytes) -> void* {
    void* p = ws + off;
    off += (bytes + 255) & ~(size_t)255;
    return p;
  };
  unsigned short* xb   = (unsigned short*)carve((size_t)N * 128 * 2);
  unsigned short* w0b  = (unsigned short*)carve((size_t)128 * 384 * 2);
  unsigned short* w1b  = (unsigned short*)carve((size_t)384 * 128 * 2);
  float*          h0   = (float*)carve((size_t)N * 384 * 4);
  float*          as0  = (float*)carve((size_t)N * 3 * 4);
  float*          ad0  = (float*)carve((size_t)N * 3 * 4);
  unsigned*       m0   = (unsigned*)carve((size_t)N * 3 * 4);
  float*          z0   = (float*)carve((size_t)N * 3 * 4);
  float*          agg0 = (float*)carve((size_t)N * 384 * 4);
  float*          sc0  = (float*)carve(384 * 4);
  float*          sh0  = (float*)carve(384 * 4);
  unsigned short* h1b  = (unsigned short*)carve((size_t)N * 384 * 2);
  float*          h1   = (float*)carve((size_t)N * 128 * 4);
  float*          as1  = (float*)carve((size_t)N * 4);
  float*          ad1  = (float*)carve((size_t)N * 4);
  unsigned*       m1   = (unsigned*)carve((size_t)N * 4);
  float*          z1   = (float*)carve((size_t)N * 4);
  float*          agg1 = (float*)carve((size_t)N * 128 * 4);
  float*          sc1  = (float*)carve(128 * 4);
  float*          sh1  = (float*)carve(128 * 4);
  float*          pooled = (float*)carve(128 * 4);

  const int rowTiles = (N + 15) / 16;          // 3125
  auto cdiv = [](long a, long b){ return (int)((a + b - 1) / b); };

  // precision staging
  k_cvt_bf16<<<cdiv((long)N * 128, 256), 256, 0, stream>>>(x, xb, (long)N * 128);
  k_pack_w<<<cdiv(128 * 384, 256), 256, 0, stream>>>(W0, w0b, 128, 384);
  k_pack_w<<<cdiv(384 * 128, 256), 256, 0, stream>>>(W1, w1b, 384, 128);

  // zero scratch accumulators (ws is poisoned between runs)
  k_zero<<<cdiv((long)N * 384, 256), 256, 0, stream>>>(agg0, (long)N * 384);
  k_zero<<<cdiv((long)N * 3, 256),   256, 0, stream>>>((float*)m0, (long)N * 3);
  k_zero<<<cdiv((long)N * 3, 256),   256, 0, stream>>>(z0, (long)N * 3);
  k_zero<<<cdiv((long)N * 128, 256), 256, 0, stream>>>(agg1, (long)N * 128);
  k_zero<<<cdiv((long)N, 256),       256, 0, stream>>>((float*)m1, (long)N);
  k_zero<<<cdiv((long)N, 256),       256, 0, stream>>>(z1, (long)N);

  // ---- layer 0: GATConv(128 -> 128, heads=3) ----
  {
    dim3 g(cdiv(rowTiles, 8), 384 / 64);
    k_gemm_bf16<<<g, 256, 0, stream>>>(xb, w0b, h0, 128, 384, rowTiles);
  }
  k_att<<<cdiv((long)N * 3, 8), 256, 0, stream>>>(h0, as0w, ad0w, as0, ad0, N, 3);
  k_edge_max<<<cdiv(ETOT, 256), 256, 0, stream>>>(ei, E, ETOT, as0, ad0, m0, 3);
  k_edge_sum<<<cdiv(ETOT, 256), 256, 0, stream>>>(ei, E, ETOT, as0, ad0, m0, z0, 3);
  k_edge_agg<<<cdiv(ETOT, 8), 256, 0, stream>>>(ei, E, ETOT, as0, ad0, m0, z0, h0, agg0, 3);
  k_bn_stats<<<384, 256, 0, stream>>>(agg0, b0, g0, be0, sc0, sh0, N, 384);
  k_bn_apply_bf16<<<cdiv((long)N * 384, 256), 256, 0, stream>>>(agg0, b0, sc0, sh0, h1b,
                                                                (long)N * 384, 384);

  // ---- layer 1: GATConv(384 -> 128, heads=1) ----
  {
    dim3 g(cdiv(rowTiles, 8), 128 / 64);
    k_gemm_bf16<<<g, 256, 0, stream>>>(h1b, w1b, h1, 384, 128, rowTiles);
  }
  k_att<<<cdiv((long)N, 8), 256, 0, stream>>>(h1, as1w, ad1w, as1, ad1, N, 1);
  k_edge_max<<<cdiv(ETOT, 256), 256, 0, stream>>>(ei, E, ETOT, as1, ad1, m1, 1);
  k_edge_sum<<<cdiv(ETOT, 256), 256, 0, stream>>>(ei, E, ETOT, as1, ad1, m1, z1, 1);
  k_edge_agg<<<cdiv(ETOT, 8), 256, 0, stream>>>(ei, E, ETOT, as1, ad1, m1, z1, h1, agg1, 1);
  k_bn_stats<<<128, 256, 0, stream>>>(agg1, b1, g1, be1, sc1, sh1, N, 128);

  // ---- pool + MLP head ----
  k_colmean<<<128, 256, 0, stream>>>(agg1, b1, sc1, sh1, pooled, N, 128);
  k_mlp<<<1, 256, 0, stream>>>(pooled, Wm1, bm1, Wm2, bm2, (float*)d_out);
}
